// DotProductAttention_55997783605811
// MI455X (gfx1250) — matrix operand order repaired and checked by reading, not compiled, not run on hardware
//
#include <hip/hip_runtime.h>

// CDNA5 / gfx1250 flash attention: B=4, H=16, S=2048, D=64, f32 in/out,
// no 1/sqrt(d) scaling (faithful to reference).
//
// v3.1: 64-key tiles (half the softmax/rescale overhead per key), exp2-domain
//     softmax (log2e folded into Q fragments -> bare v_exp_f32 via
//     __builtin_amdgcn_exp2f), packed v_cvt_pk_f16_f32 + ds_store_b32
//     V-transpose staging, DPP ROW_XMASK reductions, double-buffered LDS
//     with register-staged global loads.

typedef __attribute__((ext_vector_type(16))) _Float16 v16h;
typedef __attribute__((ext_vector_type(8)))  _Float16 v8h;
typedef __attribute__((ext_vector_type(2)))  _Float16 v2h;
typedef __attribute__((ext_vector_type(8)))  float    v8f;
typedef __attribute__((ext_vector_type(4)))  float    f4;

#define S_LEN   2048
#define D_DIM   64
#define KT      64            // keys staged per iteration
#define QW      16            // query rows per wave
#define WAVES   8
#define QB      (QW * WAVES)  // 128 query rows per block

#define KSTR    72            // LDS row stride (halves) for K tile [key][d]
#define VSTR    72            // LDS row stride (halves) for V^T tile [d][key]
#define PSTR    64            // LDS row stride (halves) for P tile [q][key]

#define LOG2E   1.4426950408889634f

__device__ __forceinline__ v16h cat16(v8h lo, v8h hi) {
    return __builtin_shufflevector(lo, hi, 0, 1, 2, 3, 4, 5, 6, 7,
                                           8, 9, 10, 11, 12, 13, 14, 15);
}

// DPP16 ROW_XMASK lane-xor within 16-lane rows (VALU pipe, no LDS traffic).
template <int M>
__device__ __forceinline__ float dpp_xor(float x) {
    int v = __builtin_amdgcn_update_dpp(0, __builtin_bit_cast(int, x),
                                        0x160 | M, 0xF, 0xF, true);
    return __builtin_bit_cast(float, v);
}
__device__ __forceinline__ float row_max16(float x) {
    x = fmaxf(x, dpp_xor<1>(x));
    x = fmaxf(x, dpp_xor<2>(x));
    x = fmaxf(x, dpp_xor<4>(x));
    x = fmaxf(x, dpp_xor<8>(x));
    return x;
}
__device__ __forceinline__ float row_sum16(float x) {
    x += dpp_xor<1>(x);
    x += dpp_xor<2>(x);
    x += dpp_xor<4>(x);
    x += dpp_xor<8>(x);
    return x;
}

__global__ void __launch_bounds__(256)
DotProductAttention_55997783605811_kernel(const float* __restrict__ Q,
                                          const float* __restrict__ K,
                                          const float* __restrict__ V,
                                          float* __restrict__ O)
{
    __shared__ _Float16 Kl[2][KT * KSTR];        // K tile, row-major [key][d]
    __shared__ _Float16 Vt[2][D_DIM * VSTR];     // V tile, transposed [d][key]
    __shared__ _Float16 Pl[WAVES][QW * PSTR];    // per-wave P tile [q][key]

    const int tid  = threadIdx.x;
    const int lane = tid & 31;
    const int wave = tid >> 5;
    const int hi   = lane >> 4;
    const int ln   = lane & 15;

    const int qblocks = S_LEN / QB;                    // 16
    const int bh      = blockIdx.x / qblocks;
    const int qblk    = blockIdx.x % qblocks;
    const int qbase   = qblk * QB + wave * QW;

    const float* Qh = Q + (size_t)bh * S_LEN * D_DIM;
    const float* Kh = K + (size_t)bh * S_LEN * D_DIM;
    const float* Vh = V + (size_t)bh * S_LEN * D_DIM;
    float*       Oh = O + (size_t)bh * S_LEN * D_DIM;

    // Staging geometry: 256 threads, each owns 2 keys x 8 dims of the 64x64 tile.
    const int kp = tid & 31;        // key pair index: keys 2kp, 2kp+1
    const int dg = tid >> 5;        // dim group 0..7
    const int d0 = dg * 8;
    const int k0 = kp * 2;

    // ---- Q tile once: A-fragment layout (16x32 f16), two chunks over D=64.
    // log2(e) folded in so softmax runs in the exp2 domain (identical result).
    v16h aq[2];
    {
        const float* qrow = Qh + (size_t)(qbase + ln) * D_DIM;
        #pragma unroll
        for (int c = 0; c < 2; ++c) {
            v16h a;
            const int dd = c * 32 + hi * 16;
            #pragma unroll
            for (int i = 0; i < 8; ++i) {
                a[i]     = (_Float16)(qrow[dd + i] * LOG2E);
                a[8 + i] = (_Float16)(qrow[dd + 8 + i] * LOG2E);
            }
            aq[c] = a;
        }
    }

    float mrow[8], lrow[8];
    v8f   o[4];
    #pragma unroll
    for (int r = 0; r < 8; ++r) { mrow[r] = -3.0e38f; lrow[r] = 0.0f; }
    #pragma unroll
    for (int c = 0; c < 4; ++c)
        #pragma unroll
        for (int r = 0; r < 8; ++r) o[c][r] = 0.0f;

    // ---- Stage one 64-key tile (f32 regs -> f16 LDS).
    auto stage = [&](int buf, const f4* kreg, const f4* vreg) {
        v8h h0, h1;
        #pragma unroll
        for (int i = 0; i < 4; ++i) {
            h0[i] = (_Float16)kreg[0][i];  h0[4 + i] = (_Float16)kreg[1][i];
            h1[i] = (_Float16)kreg[2][i];  h1[4 + i] = (_Float16)kreg[3][i];
        }
        *(v8h*)&Kl[buf][(k0 + 0) * KSTR + d0] = h0;
        *(v8h*)&Kl[buf][(k0 + 1) * KSTR + d0] = h1;
        #pragma unroll
        for (int i = 0; i < 8; ++i) {
            v2h p;                           // (key k0, key k0+1) packed pair
            p[0] = (_Float16)vreg[i >> 2][i & 3];
            p[1] = (_Float16)vreg[2 + (i >> 2)][i & 3];
            *(v2h*)&Vt[buf][(d0 + i) * VSTR + k0] = p;
        }
    };
    auto load_tile = [&](int kb, f4* kreg, f4* vreg) {
        const float* ks0 = Kh + (size_t)(kb + k0) * D_DIM + d0;
        const float* ks1 = ks0 + D_DIM;
        const float* vs0 = Vh + (size_t)(kb + k0) * D_DIM + d0;
        const float* vs1 = vs0 + D_DIM;
        kreg[0] = *(const f4*)ks0;  kreg[1] = *(const f4*)(ks0 + 4);
        kreg[2] = *(const f4*)ks1;  kreg[3] = *(const f4*)(ks1 + 4);
        vreg[0] = *(const f4*)vs0;  vreg[1] = *(const f4*)(vs0 + 4);
        vreg[2] = *(const f4*)vs1;  vreg[3] = *(const f4*)(vs1 + 4);
    };

    // ---- Prologue: stage tile 0 into buffer 0.
    {
        f4 kreg[4], vreg[4];
        load_tile(0, kreg, vreg);
        stage(0, kreg, vreg);
    }
    __syncthreads();

    const int NIT = S_LEN / KT;     // 32 key-tile iterations
    for (int kt = 0; kt < NIT; ++kt) {
        const int cur = kt & 1;
        const int nxt = cur ^ 1;
        const bool have_next = (kt + 1) < NIT;

        // Issue next tile's global loads early (hidden behind WMMA + softmax).
        f4 kreg[4], vreg[4];
        if (have_next) load_tile((kt + 1) * KT, kreg, vreg);

        // ---- S = Q * K^T : four 16-key sub-tiles, K-dim 64 = 2 x 32.
        v8f sc[4];
        #pragma unroll
        for (int j = 0; j < 4; ++j) {
            v16h b0, b1;
            {
                const _Float16* bp = &Kl[cur][(j * 16 + ln) * KSTR + hi * 16];
                b0 = cat16(*(const v8h*)bp, *(const v8h*)(bp + 8));
                b1 = cat16(*(const v8h*)(bp + 32), *(const v8h*)(bp + 40));
            }
            v8f acc;
            #pragma unroll
            for (int i = 0; i < 8; ++i) acc[i] = 0.0f;
            acc = __builtin_amdgcn_wmma_f32_16x16x32_f16(false, aq[0], false, b0,
                                                         (short)0, acc, false, false);
            acc = __builtin_amdgcn_wmma_f32_16x16x32_f16(false, aq[1], false, b1,
                                                         (short)0, acc, false, false);
            sc[j] = acc;
        }

        // ---- Online softmax over 64 keys (exp2 domain, DPP row reductions).
        float alpha[8];
        #pragma unroll
        for (int r = 0; r < 8; ++r) {
            float tm = fmaxf(fmaxf(sc[0][r], sc[1][r]), fmaxf(sc[2][r], sc[3][r]));
            tm = row_max16(tm);
            const float nm = fmaxf(mrow[r], tm);
            alpha[r] = __builtin_amdgcn_exp2f(mrow[r] - nm);
            mrow[r]  = nm;
            float rs = 0.0f;
            #pragma unroll
            for (int j = 0; j < 4; ++j) {
                const float p = __builtin_amdgcn_exp2f(sc[j][r] - nm);
                sc[j][r] = p;
                rs += p;
            }
            lrow[r] = lrow[r] * alpha[r] + row_sum16(rs);
        }
        #pragma unroll
        for (int c = 0; c < 4; ++c)
            #pragma unroll
            for (int r = 0; r < 8; ++r) o[c][r] *= alpha[r];

        // ---- C-layout P -> LDS -> A-layout fragments (wave-local).
        _Float16* prow = &Pl[wave][0];
        #pragma unroll
        for (int j = 0; j < 4; ++j)
            #pragma unroll
            for (int r = 0; r < 8; ++r)
                prow[(hi * 8 + r) * PSTR + j * 16 + ln] = (_Float16)sc[j][r];

        asm volatile("s_wait_dscnt 0" ::: "memory");

        v16h pf[2];
        #pragma unroll
        for (int kc = 0; kc < 2; ++kc) {
            const _Float16* pp = prow + ln * PSTR + kc * 32 + hi * 16;
            pf[kc] = cat16(*(const v8h*)pp, *(const v8h*)(pp + 8));
        }

        // ---- O += P * V : K-dim 64 keys = 2 chunks of 32; 4 output col chunks.
        #pragma unroll
        for (int kc = 0; kc < 2; ++kc) {
            v16h bv[4];
            #pragma unroll
            for (int c = 0; c < 4; ++c) {
                const _Float16* vp = &Vt[cur][(c * 16 + ln) * VSTR + kc * 32 + hi * 16];
                bv[c] = cat16(*(const v8h*)vp, *(const v8h*)(vp + 8));
            }
            #pragma unroll
            for (int c = 0; c < 4; ++c)
                o[c] = __builtin_amdgcn_wmma_f32_16x16x32_f16(
                           false, pf[kc], false, bv[c],
                           (short)0, o[c], false, false);
        }

        // ---- Convert + store next tile into the alternate LDS buffer.
        if (have_next) stage(nxt, kreg, vreg);
        __syncthreads();
    }

    // ---- Epilogue: divide by row sums, store f32.
    #pragma unroll
    for (int r = 0; r < 8; ++r) lrow[r] = 1.0f / lrow[r];
    #pragma unroll
    for (int c = 0; c < 4; ++c) {
        #pragma unroll
        for (int r = 0; r < 8; ++r) {
            const int row = qbase + hi * 8 + r;
            Oh[(size_t)row * D_DIM + c * 16 + ln] = o[c][r] * lrow[r];
        }
    }
}

extern "C" void kernel_launch(void* const* d_in, const int* in_sizes, int n_in,
                              void* d_out, int out_size, void* d_ws, size_t ws_size,
                              hipStream_t stream) {
    const float* Q = (const float*)d_in[0];
    const float* K = (const float*)d_in[1];
    const float* V = (const float*)d_in[2];
    float*       O = (float*)d_out;

    const int BH = 4 * 16;                 // B*H
    dim3 grid(BH * (S_LEN / QB));          // 1024 blocks
    dim3 block(256);                       // 8 waves (wave32)
    hipLaunchKernelGGL(DotProductAttention_55997783605811_kernel,
                       grid, block, 0, stream, Q, K, V, O);
}
